// DynamicMambaSSM_21904333209604
// MI455X (gfx1250) — compile-verified, matrix operand order checked by reading, not run on hardware
//
#include <hip/hip_runtime.h>

#define D_MODEL   1024
#define D_INNER   2048
#define D_STATE   128
#define NHEADS    32
#define HEADDIM   64
#define D_CONV    4
#define CHUNK     256
#define CONV_DIM  (D_INNER + 2 * D_STATE)              // 2304
#define D_IN_PROJ (2 * D_INNER + 2 * D_STATE + NHEADS) // 4384
#define BATCH     2
#define SEQLEN    4096
#define NCHUNK    (SEQLEN / CHUNK)                     // 16
#define NTOK      (BATCH * SEQLEN)                     // 8192

typedef __bf16 bf16;
typedef __attribute__((ext_vector_type(16))) __bf16 v16bf;
typedef __attribute__((ext_vector_type(8)))  float  v8f;
typedef __attribute__((ext_vector_type(2)))  int    v2i;

// ---------------------------------------------------------------------------
// Optional CDNA5 async global->LDS copy (ASYNCcnt-tracked). Guarded so the
// code still compiles (falling back to VGPR staging) if the builtin is not
// declared by this toolchain. Probe showed param0 is a global-AS v2i*.
// ---------------------------------------------------------------------------
#if defined(__has_builtin)
#if __has_builtin(__builtin_amdgcn_global_load_async_to_lds_b64) && \
    __has_builtin(__builtin_amdgcn_s_wait_asynccnt)
#define ASYNC_CP 1
#endif
#endif

typedef __attribute__((address_space(1))) v2i gas_v2i;
typedef __attribute__((address_space(3))) v2i las_v2i;

__device__ __forceinline__ void cp_b64(bf16* lds_dst, const bf16* gsrc) {
#ifdef ASYNC_CP
  __builtin_amdgcn_global_load_async_to_lds_b64(
      (gas_v2i*)gsrc, (las_v2i*)lds_dst, 0, 0);
#else
  *(uint2*)lds_dst = *(const uint2*)gsrc;
#endif
}

__device__ __forceinline__ void cp_wait() {
#ifdef ASYNC_CP
  __builtin_amdgcn_s_wait_asynccnt(0);
#endif
}

// ---------------------------------------------------------------------------
// WMMA helpers (v_wmma_f32_16x16x32_bf16, wave32)
// A fragment: 16(M) x 32(K) bf16 from LDS, row-major, stride `ld` elements.
//   lane<16 : row=lane,    elems 0..7 = K 0..7,  elems 8..15 = K 16..23
//   lane>=16: row=lane-16, elems 0..7 = K 8..15, elems 8..15 = K 24..31
// B fragment: same layout with N in place of M (B supplied as [N][K]).
// ---------------------------------------------------------------------------
__device__ __forceinline__ v16bf frag_ld(const bf16* base, int ld) {
  const int lane = threadIdx.x & 31;
  const bf16* p = base + (lane & 15) * ld + ((lane >> 4) << 3);
  union { v16bf v; uint4 u[2]; } f;
  f.u[0] = *(const uint4*)(p);
  f.u[1] = *(const uint4*)(p + 16);
  return f.v;
}

__device__ __forceinline__ v8f wmma_bf(v16bf a, v16bf b, v8f c) {
  return __builtin_amdgcn_wmma_f32_16x16x32_bf16(false, a, false, b,
                                                 (short)0, c, false, false);
}

__device__ __forceinline__ float siluf(float x) {
  return x / (1.0f + __expf(-x));
}

// ---------------------------------------------------------------------------
// Generic (batched) bf16 WMMA GEMM:  C[M,N] = A[M,K] * B[N,K]^T
// A row-major [M,K]; B row-major [N,K] (i.e. already transposed operand).
// AF32/BF32: source is f32 (converted to bf16 while staging into LDS);
// bf16 sources are staged with async global->LDS copies when available.
// ---------------------------------------------------------------------------
template <bool AF32, bool BF32>
__global__ __launch_bounds__(256) void gemm_k(
    const void* __restrict__ Ap, const void* __restrict__ Bp,
    float* __restrict__ Cp, int M, int N, int K,
    long strideA, long strideB, long strideC)
{
  constexpr int BM = 128, BN = 128, BK = 32, LD = 40;
  __shared__ bf16 As[BM * LD];
  __shared__ bf16 Bs[BN * LD];

  const int tid = threadIdx.x;
  const int m0 = blockIdx.y * BM, n0 = blockIdx.x * BN;
  const size_t z = blockIdx.z;

  const float* Af = nullptr; const bf16* Ah = nullptr;
  const float* Bf = nullptr; const bf16* Bh = nullptr;
  if constexpr (AF32) Af = (const float*)Ap + z * strideA;
  else                Ah = (const bf16*)Ap + z * strideA;
  if constexpr (BF32) Bf = (const float*)Bp + z * strideB;
  else                Bh = (const bf16*)Bp + z * strideB;
  float* C = Cp + z * strideC;

  const int wid = tid >> 5;
  const int wm = wid & 3;   // wave rows: wm*32 (two 16-row tiles)
  const int wn = wid >> 2;  // wave cols: wn*64 (four 16-col tiles)

  v8f acc[2][4] = {};

  for (int k0 = 0; k0 < K; k0 += BK) {
    const int r = (tid >> 3);
    const int c4 = (tid & 7) * 4;
    // prefetch next K tile (lowers to global_prefetch_b8)
    if (k0 + BK < K) {
      if constexpr (AF32) {
        if (m0 + r < M) __builtin_prefetch(Af + (size_t)(m0 + r) * K + k0 + BK + c4, 0, 0);
      } else {
        if (m0 + r < M) __builtin_prefetch(Ah + (size_t)(m0 + r) * K + k0 + BK + c4, 0, 0);
      }
      if constexpr (BF32) {
        if (n0 + r < N) __builtin_prefetch(Bf + (size_t)(n0 + r) * K + k0 + BK + c4, 0, 0);
      } else {
        if (n0 + r < N) __builtin_prefetch(Bh + (size_t)(n0 + r) * K + k0 + BK + c4, 0, 0);
      }
    }
    // stage A tile (BM x BK)
#pragma unroll
    for (int i = 0; i < 4; i++) {
      int rr = r + i * 32;
      bf16* d = &As[rr * LD + c4];
      int gr = m0 + rr;
      if (gr < M) {
        if constexpr (AF32) {
          float4 v = *(const float4*)(Af + (size_t)gr * K + k0 + c4);
          d[0] = (bf16)v.x; d[1] = (bf16)v.y; d[2] = (bf16)v.z; d[3] = (bf16)v.w;
        } else {
          cp_b64(d, Ah + (size_t)gr * K + k0 + c4);
        }
      } else {
        d[0] = (bf16)0.f; d[1] = (bf16)0.f; d[2] = (bf16)0.f; d[3] = (bf16)0.f;
      }
    }
    // stage B tile (BN x BK)
#pragma unroll
    for (int i = 0; i < 4; i++) {
      int rr = r + i * 32;
      bf16* d = &Bs[rr * LD + c4];
      int gr = n0 + rr;
      if (gr < N) {
        if constexpr (BF32) {
          float4 v = *(const float4*)(Bf + (size_t)gr * K + k0 + c4);
          d[0] = (bf16)v.x; d[1] = (bf16)v.y; d[2] = (bf16)v.z; d[3] = (bf16)v.w;
        } else {
          cp_b64(d, Bh + (size_t)gr * K + k0 + c4);
        }
      } else {
        d[0] = (bf16)0.f; d[1] = (bf16)0.f; d[2] = (bf16)0.f; d[3] = (bf16)0.f;
      }
    }
    cp_wait();
    __syncthreads();
    v16bf af[2], bfrag[4];
#pragma unroll
    for (int i = 0; i < 2; i++) af[i] = frag_ld(&As[(wm * 32 + i * 16) * LD], LD);
#pragma unroll
    for (int j = 0; j < 4; j++) bfrag[j] = frag_ld(&Bs[(wn * 64 + j * 16) * LD], LD);
#pragma unroll
    for (int i = 0; i < 2; i++)
#pragma unroll
      for (int j = 0; j < 4; j++)
        acc[i][j] = wmma_bf(af[i], bfrag[j], acc[i][j]);
    __syncthreads();
  }

  const int lane = tid & 31;
  const int ccol = lane & 15;
  const int rsel = (lane >> 4) << 3;
#pragma unroll
  for (int i = 0; i < 2; i++)
#pragma unroll
    for (int j = 0; j < 4; j++) {
      int col = n0 + wn * 64 + j * 16 + ccol;
      int rb = m0 + wm * 32 + i * 16 + rsel;
#pragma unroll
      for (int e = 0; e < 8; e++) {
        int row = rb + e;
        if (row < M && col < N) C[(size_t)row * N + col] = acc[i][j][e];
      }
    }
}

// ---------------------------------------------------------------------------
// Depthwise causal conv(4) + SiLU, producing f32 x and bf16 staging copies.
// ---------------------------------------------------------------------------
__global__ __launch_bounds__(256) void conv_silu_k(
    const float* __restrict__ zxbcdt, const float* __restrict__ conv_w,
    const float* __restrict__ conv_b,
    float* __restrict__ xf32, bf16* __restrict__ xT,
    bf16* __restrict__ Bc, bf16* __restrict__ BcT, bf16* __restrict__ Cc)
{
  const int c = blockIdx.x * 256 + threadIdx.x; // 0..CONV_DIM-1
  const int l = blockIdx.y;
  const int b = blockIdx.z;

  float acc = conv_b[c];
#pragma unroll
  for (int k = 0; k < D_CONV; k++) {
    int ls = l - (D_CONV - 1) + k;
    if (ls >= 0)
      acc += conv_w[c * D_CONV + k] *
             zxbcdt[(size_t)(b * SEQLEN + ls) * D_IN_PROJ + D_INNER + c];
  }
  float s = siluf(acc);

  const int ch = l >> 8;     // chunk
  const int t = l & 255;     // pos in chunk
  const size_t bc = (size_t)b * NCHUNK + ch;

  if (c < D_INNER) {
    xf32[(size_t)(b * SEQLEN + l) * D_INNER + c] = s;
    int h = c >> 6, p = c & 63;
    xT[((bc * NHEADS + h) * HEADDIM + p) * CHUNK + t] = (bf16)s;
  } else if (c < D_INNER + D_STATE) {
    int n = c - D_INNER;
    Bc[(bc * CHUNK + t) * D_STATE + n] = (bf16)s;
    BcT[(bc * D_STATE + n) * CHUNK + t] = (bf16)s;
  } else {
    int n = c - D_INNER - D_STATE;
    Cc[(bc * CHUNK + t) * D_STATE + n] = (bf16)s;
  }
}

// ---------------------------------------------------------------------------
// dt = softplus(dt_raw + dt_mod + dt_bias);  a = dt * (-exp(A_log));
// cA = inclusive cumsum of a over the chunk (Hillis-Steele in LDS).
// grid (NHEADS, NCHUNK, BATCH), 256 threads = CHUNK.
// ---------------------------------------------------------------------------
__global__ __launch_bounds__(256) void dt_cumsum_k(
    const float* __restrict__ zxbcdt, const float* __restrict__ dt_mod,
    const float* __restrict__ dt_bias, const float* __restrict__ A_log,
    float* __restrict__ dts, float* __restrict__ cA)
{
  const int h = blockIdx.x, c = blockIdx.y, b = blockIdx.z;
  const int t = threadIdx.x;
  const int l = c * CHUNK + t;

  float v = zxbcdt[(size_t)(b * SEQLEN + l) * D_IN_PROJ + D_INNER + CONV_DIM + h]
          + dt_mod[(size_t)(b * SEQLEN + l) * NHEADS + h] + dt_bias[h];
  float dt = (v > 20.f) ? v : log1pf(__expf(v));
  float a = dt * (-__expf(A_log[h]));

  __shared__ float sc[CHUNK];
  sc[t] = a;
  __syncthreads();
  for (int off = 1; off < CHUNK; off <<= 1) {
    float add = (t >= off) ? sc[t - off] : 0.f;
    __syncthreads();
    sc[t] += add;
    __syncthreads();
  }
  size_t o = (((size_t)b * NCHUNK + c) * NHEADS + h) * CHUNK + t;
  dts[o] = dt;
  cA[o] = sc[t];
}

// ---------------------------------------------------------------------------
// y_intra: per (b,c,h)  y[t,p] = sum_s M[t,s]*xT[p,s] + D[h]*x[t,p]
//   M[t,s] = CB[t,s]*exp(cA[t]-cA[s])*dt[s]  (s<=t)   built panelwise in LDS.
// grid (NHEADS, NCHUNK, BATCH).
// ---------------------------------------------------------------------------
__global__ __launch_bounds__(256) void yintra_k(
    const float* __restrict__ CB, const float* __restrict__ cA,
    const float* __restrict__ dts, const bf16* __restrict__ xT,
    const float* __restrict__ xf32, const float* __restrict__ Dv,
    float* __restrict__ y)
{
  constexpr int LD = 40;
  __shared__ bf16 As[CHUNK * LD];
  __shared__ bf16 Bs[HEADDIM * LD];
  __shared__ float cAs[CHUNK], dtl[CHUNK];

  const int h = blockIdx.x, c = blockIdx.y, b = blockIdx.z;
  const int tid = threadIdx.x;
  const size_t bch = ((size_t)b * NCHUNK + c) * NHEADS + h;

  cAs[tid] = cA[bch * CHUNK + tid];
  dtl[tid] = dts[bch * CHUNK + tid];

  const float* CBp = CB + ((size_t)b * NCHUNK + c) * CHUNK * CHUNK;
  const bf16* xTp = xT + bch * HEADDIM * CHUNK; // [p][t]

  const int wid = tid >> 5;
  v8f acc[2][4] = {};
  __syncthreads();

  for (int k0 = 0; k0 < CHUNK; k0 += 32) {
    const int ssub = tid & 31;
    const int s = k0 + ssub;
#pragma unroll
    for (int i = 0; i < 32; i++) {           // M panel: 256 rows x 32 cols
      int r = i * 8 + (tid >> 5);
      float m = 0.f;
      if (s <= r) m = CBp[(size_t)r * CHUNK + s] *
                      __expf(cAs[r] - cAs[s]) * dtl[s];
      As[r * LD + ssub] = (bf16)m;
    }
    // xT panel: 64 x 32, 8-byte chunked copy (async if available)
#pragma unroll
    for (int i = 0; i < 2; i++) {
      int idx = tid + i * 256;
      int p = idx >> 3;
      int c4 = (idx & 7) * 4;
      cp_b64(&Bs[p * LD + c4], xTp + (size_t)p * CHUNK + k0 + c4);
    }
    cp_wait();
    __syncthreads();
    v16bf af[2], bfrag[4];
    af[0] = frag_ld(&As[(wid * 32) * LD], LD);
    af[1] = frag_ld(&As[(wid * 32 + 16) * LD], LD);
#pragma unroll
    for (int j = 0; j < 4; j++) bfrag[j] = frag_ld(&Bs[(j * 16) * LD], LD);
#pragma unroll
    for (int i = 0; i < 2; i++)
#pragma unroll
      for (int j = 0; j < 4; j++)
        acc[i][j] = wmma_bf(af[i], bfrag[j], acc[i][j]);
    __syncthreads();
  }

  const float Dh = Dv[h];
  const int lane = tid & 31;
  const int ccol = lane & 15;
  const int rsel = (lane >> 4) << 3;
#pragma unroll
  for (int i = 0; i < 2; i++)
#pragma unroll
    for (int j = 0; j < 4; j++)
#pragma unroll
      for (int e = 0; e < 8; e++) {
        int t = wid * 32 + i * 16 + rsel + e;
        int p = j * 16 + ccol;
        size_t l = (size_t)b * SEQLEN + c * CHUNK + t;
        size_t idx = l * D_INNER + h * HEADDIM + p;
        y[idx] = acc[i][j][e] + Dh * xf32[idx];
      }
}

// ---------------------------------------------------------------------------
// states: per (b,c,h)  states[p,n] = sum_s w[s]*xT[p,s]*BcT[n,s]
//   w[s] = exp(cA[255]-cA[s]) * dt[s]
// ---------------------------------------------------------------------------
__global__ __launch_bounds__(256) void states_k(
    const float* __restrict__ cA, const float* __restrict__ dts,
    const bf16* __restrict__ xT, const bf16* __restrict__ BcT,
    float* __restrict__ states)
{
  constexpr int LD = 40;
  __shared__ bf16 As[HEADDIM * LD];
  __shared__ bf16 Bs[D_STATE * LD];
  __shared__ float ws[CHUNK];
  __shared__ float calast;

  const int h = blockIdx.x, c = blockIdx.y, b = blockIdx.z;
  const int tid = threadIdx.x;
  const size_t bch = ((size_t)b * NCHUNK + c) * NHEADS + h;
  const size_t bc = (size_t)b * NCHUNK + c;

  float ca = cA[bch * CHUNK + tid];
  if (tid == CHUNK - 1) calast = ca;
  __syncthreads();
  ws[tid] = __expf(calast - ca) * dts[bch * CHUNK + tid];

  const bf16* xTp = xT + bch * HEADDIM * CHUNK; // [p][t]
  const bf16* Bp = BcT + bc * D_STATE * CHUNK;  // [n][t]

  const int wid = tid >> 5;
  const int wm = wid & 3;   // rows p: wm*16
  const int wn = wid >> 2;  // cols n: wn*64
  v8f acc[4] = {};
  __syncthreads();

  for (int k0 = 0; k0 < CHUNK; k0 += 32) {
    const int ssub = tid & 31;
    const int s = k0 + ssub;
    // weighted xT panel: 64 x 32 (needs VALU scaling)
#pragma unroll
    for (int i = 0; i < 8; i++) {
      int p = i * 8 + (tid >> 5);
      As[p * LD + ssub] = (bf16)((float)xTp[(size_t)p * CHUNK + s] * ws[s]);
    }
    // BcT panel: 128 x 32, 8-byte chunked copy (async if available)
#pragma unroll
    for (int i = 0; i < 4; i++) {
      int idx = tid + i * 256;
      int n = idx >> 3;
      int c4 = (idx & 7) * 4;
      cp_b64(&Bs[n * LD + c4], Bp + (size_t)n * CHUNK + k0 + c4);
    }
    cp_wait();
    __syncthreads();
    v16bf a = frag_ld(&As[(wm * 16) * LD], LD);
#pragma unroll
    for (int j = 0; j < 4; j++) {
      v16bf bfrag = frag_ld(&Bs[(wn * 64 + j * 16) * LD], LD);
      acc[j] = wmma_bf(a, bfrag, acc[j]);
    }
    __syncthreads();
  }

  const int lane = tid & 31;
  const int ccol = lane & 15;
  const int rsel = (lane >> 4) << 3;
  float* So = states + bch * HEADDIM * D_STATE;
#pragma unroll
  for (int j = 0; j < 4; j++)
#pragma unroll
    for (int e = 0; e < 8; e++) {
      int p = wm * 16 + rsel + e;
      int n = wn * 64 + j * 16 + ccol;
      So[(size_t)p * D_STATE + n] = acc[j][e];
    }
}

// ---------------------------------------------------------------------------
// Inter-chunk sequential scan: S_in[c] = S;  S = exp(cA[c,last])*S + states[c]
// grid (NHEADS, BATCH), 256 threads; each thread owns 32 state elements.
// ---------------------------------------------------------------------------
__global__ __launch_bounds__(256) void scan_k(
    const float* __restrict__ cA, const float* __restrict__ states,
    bf16* __restrict__ Sin)
{
  const int h = blockIdx.x, b = blockIdx.y;
  const int tid = threadIdx.x;
  float S[32];
#pragma unroll
  for (int j = 0; j < 32; j++) S[j] = 0.f;

  for (int c = 0; c < NCHUNK; c++) {
    size_t bch = ((size_t)b * NCHUNK + c) * NHEADS + h;
    float dec = __expf(cA[bch * CHUNK + (CHUNK - 1)]);
    const float* st = states + bch * HEADDIM * D_STATE;
    bf16* so = Sin + bch * HEADDIM * D_STATE;
#pragma unroll
    for (int j = 0; j < 32; j++) {
      int idx = j * 256 + tid;
      so[idx] = (bf16)S[j];
      S[j] = dec * S[j] + st[idx];
    }
  }
}

// ---------------------------------------------------------------------------
// y_inter: per (b,c,h)  y[t,p] += sum_n exp(cA[t])*Cc[t,n]*S_in[p,n]
// ---------------------------------------------------------------------------
__global__ __launch_bounds__(256) void yinter_k(
    const float* __restrict__ cA, const bf16* __restrict__ Cc,
    const bf16* __restrict__ Sin, float* __restrict__ y)
{
  constexpr int LD = 40;
  __shared__ bf16 As[CHUNK * LD];
  __shared__ bf16 Bs[HEADDIM * LD];
  __shared__ float es[CHUNK];

  const int h = blockIdx.x, c = blockIdx.y, b = blockIdx.z;
  const int tid = threadIdx.x;
  const size_t bch = ((size_t)b * NCHUNK + c) * NHEADS + h;
  const size_t bc = (size_t)b * NCHUNK + c;

  es[tid] = __expf(cA[bch * CHUNK + tid]);
  const bf16* Cp = Cc + bc * CHUNK * D_STATE;     // [t][n]
  const bf16* Sp = Sin + bch * HEADDIM * D_STATE; // [p][n]

  const int wid = tid >> 5;
  v8f acc[2][4] = {};
  __syncthreads();

  for (int k0 = 0; k0 < D_STATE; k0 += 32) {
    const int nsub = tid & 31;
    const int n = k0 + nsub;
#pragma unroll
    for (int i = 0; i < 32; i++) {           // scaled Cc panel: 256 x 32
      int r = i * 8 + (tid >> 5);
      As[r * LD + nsub] = (bf16)((float)Cp[(size_t)r * D_STATE + n] * es[r]);
    }
    // S_in panel: 64 x 32, 8-byte chunked copy (async if available)
#pragma unroll
    for (int i = 0; i < 2; i++) {
      int idx = tid + i * 256;
      int p = idx >> 3;
      int c4 = (idx & 7) * 4;
      cp_b64(&Bs[p * LD + c4], Sp + (size_t)p * D_STATE + k0 + c4);
    }
    cp_wait();
    __syncthreads();
    v16bf af[2], bfrag[4];
    af[0] = frag_ld(&As[(wid * 32) * LD], LD);
    af[1] = frag_ld(&As[(wid * 32 + 16) * LD], LD);
#pragma unroll
    for (int j = 0; j < 4; j++) bfrag[j] = frag_ld(&Bs[(j * 16) * LD], LD);
#pragma unroll
    for (int i = 0; i < 2; i++)
#pragma unroll
      for (int j = 0; j < 4; j++)
        acc[i][j] = wmma_bf(af[i], bfrag[j], acc[i][j]);
    __syncthreads();
  }

  const int lane = tid & 31;
  const int ccol = lane & 15;
  const int rsel = (lane >> 4) << 3;
#pragma unroll
  for (int i = 0; i < 2; i++)
#pragma unroll
    for (int j = 0; j < 4; j++)
#pragma unroll
      for (int e = 0; e < 8; e++) {
        int t = wid * 32 + i * 16 + rsel + e;
        int p = j * 16 + ccol;
        size_t l = (size_t)b * SEQLEN + c * CHUNK + t;
        size_t idx = l * D_INNER + h * HEADDIM + p;
        y[idx] += acc[i][j][e];
      }
}

// ---------------------------------------------------------------------------
// Gating (y * silu(z)) + RMSNorm -> bf16 for out_proj.
// grid (SEQLEN, BATCH), 256 threads, 8 channels/thread.
// ---------------------------------------------------------------------------
__global__ __launch_bounds__(256) void gatenorm_k(
    const float* __restrict__ zxbcdt, const float* __restrict__ y,
    const float* __restrict__ norm_w, bf16* __restrict__ ynorm)
{
  const int l = blockIdx.x, b = blockIdx.y;
  const int tid = threadIdx.x;
  const size_t row = (size_t)b * SEQLEN + l;
  const float* zp = zxbcdt + row * D_IN_PROJ;
  const float* yp = y + row * D_INNER;

  float vals[8];
  float ss = 0.f;
#pragma unroll
  for (int i = 0; i < 8; i++) {
    int cidx = tid + i * 256;
    float z = zp[cidx];
    float g = yp[cidx] * siluf(z);
    vals[i] = g;
    ss += g * g;
  }
  __shared__ float red[256];
  red[tid] = ss;
  __syncthreads();
  for (int off = 128; off > 0; off >>= 1) {
    if (tid < off) red[tid] += red[tid + off];
    __syncthreads();
  }
  float rstd = rsqrtf(red[0] / (float)D_INNER + 1e-5f);
#pragma unroll
  for (int i = 0; i < 8; i++) {
    int cidx = tid + i * 256;
    ynorm[row * D_INNER + cidx] = (bf16)(vals[i] * rstd * norm_w[cidx]);
  }
}

// ---------------------------------------------------------------------------
extern "C" void kernel_launch(void* const* d_in, const int* in_sizes, int n_in,
                              void* d_out, int out_size, void* d_ws, size_t ws_size,
                              hipStream_t stream) {
  const float* u       = (const float*)d_in[0];
  const float* dt_mod  = (const float*)d_in[1];
  const float* W_in    = (const float*)d_in[2];
  const float* conv_w  = (const float*)d_in[3];
  const float* conv_b  = (const float*)d_in[4];
  const float* dt_bias = (const float*)d_in[5];
  const float* A_log   = (const float*)d_in[6];
  const float* Dv      = (const float*)d_in[7];
  const float* norm_w  = (const float*)d_in[8];
  const float* W_out   = (const float*)d_in[9];
  float* out = (float*)d_out;

  char* ws = (char*)d_ws;
  auto alloc = [&](size_t bytes) {
    char* p = ws;
    ws += (bytes + 255) & ~(size_t)255;
    return (void*)p;
  };

  float* zxbcdt = (float*)alloc((size_t)NTOK * D_IN_PROJ * 4);
  float* xf32   = (float*)alloc((size_t)NTOK * D_INNER * 4);
  float* ybuf   = (float*)alloc((size_t)NTOK * D_INNER * 4);
  bf16* xT      = (bf16*)alloc((size_t)NTOK * D_INNER * 2);
  bf16* Bc      = (bf16*)alloc((size_t)BATCH * NCHUNK * CHUNK * D_STATE * 2);
  bf16* BcT     = (bf16*)alloc((size_t)BATCH * NCHUNK * CHUNK * D_STATE * 2);
  bf16* Cc      = (bf16*)alloc((size_t)BATCH * NCHUNK * CHUNK * D_STATE * 2);
  float* dts    = (float*)alloc((size_t)BATCH * NCHUNK * NHEADS * CHUNK * 4);
  float* cAb    = (float*)alloc((size_t)BATCH * NCHUNK * NHEADS * CHUNK * 4);
  float* CB     = (float*)alloc((size_t)BATCH * NCHUNK * CHUNK * CHUNK * 4);
  float* states = (float*)alloc((size_t)BATCH * NCHUNK * NHEADS * HEADDIM * D_STATE * 4);
  bf16* Sin     = (bf16*)alloc((size_t)BATCH * NCHUNK * NHEADS * HEADDIM * D_STATE * 2);
  // ynorm aliases the states buffer: states is fully consumed by scan_k
  // before gatenorm_k runs (stream-ordered), and sizes match exactly.
  bf16* ynorm = (bf16*)states;

  // 1. in_proj: zxbcdt[8192,4384] = u[8192,1024] @ W_in[4384,1024]^T
  gemm_k<true, true><<<dim3((D_IN_PROJ + 127) / 128, NTOK / 128, 1), 256, 0, stream>>>(
      u, W_in, zxbcdt, NTOK, D_IN_PROJ, D_MODEL, 0, 0, 0);

  // 2. conv + silu + staging copies
  conv_silu_k<<<dim3(CONV_DIM / 256, SEQLEN, BATCH), 256, 0, stream>>>(
      zxbcdt, conv_w, conv_b, xf32, xT, Bc, BcT, Cc);

  // 3. dt softplus + cumsum
  dt_cumsum_k<<<dim3(NHEADS, NCHUNK, BATCH), 256, 0, stream>>>(
      zxbcdt, dt_mod, dt_bias, A_log, dts, cAb);

  // 4. CB[bc,256,256] = Cc @ Bc^T (batched)
  gemm_k<false, false><<<dim3(2, 2, BATCH * NCHUNK), 256, 0, stream>>>(
      Cc, Bc, CB, CHUNK, CHUNK, D_STATE,
      (long)CHUNK * D_STATE, (long)CHUNK * D_STATE, (long)CHUNK * CHUNK);

  // 5. intra-chunk output + skip connection
  yintra_k<<<dim3(NHEADS, NCHUNK, BATCH), 256, 0, stream>>>(
      CB, cAb, dts, xT, xf32, Dv, ybuf);

  // 6. chunk-local states
  states_k<<<dim3(NHEADS, NCHUNK, BATCH), 256, 0, stream>>>(
      cAb, dts, xT, BcT, states);

  // 7. inter-chunk scan -> S_in (bf16)
  scan_k<<<dim3(NHEADS, BATCH), 256, 0, stream>>>(cAb, states, Sin);

  // 8. inter-chunk contribution accumulated into y
  yinter_k<<<dim3(NHEADS, NCHUNK, BATCH), 256, 0, stream>>>(cAb, Cc, Sin, ybuf);

  // 9. gating + RMSNorm -> bf16
  gatenorm_k<<<dim3(SEQLEN, BATCH), 256, 0, stream>>>(zxbcdt, ybuf, norm_w, ynorm);

  // 10. out_proj: out[8192,1024] = ynorm[8192,2048] @ W_out[1024,2048]^T
  gemm_k<false, true><<<dim3(D_MODEL / 128, NTOK / 128, 1), 256, 0, stream>>>(
      ynorm, W_out, out, NTOK, D_MODEL, D_INNER, 0, 0, 0);
}
